// VRNNCell_69552700391572
// MI455X (gfx1250) — matrix operand order repaired and checked by reading, not compiled, not run on hardware
//
#include <hip/hip_runtime.h>
#include <hip/hip_bf16.h>
#include <math.h>

// ---------------------------------------------------------------------------
// VRNN step for MI455X (gfx1250, wave32).
//  * all GEMMs on v_wmma_f32_16x16x32_bf16 (fp32 accumulate)
//  * fp32 weights/activations pre-packed to bf16 once (bandwidth halved,
//    weights live in the 192MB L2 across all 8 GEMMs)
//  * tiles staged global->LDS by the Tensor Data Mover (tensor_load_to_lds)
//    with hardware LDS row padding; double buffered; s_wait_tensorcnt sync
// ---------------------------------------------------------------------------

typedef __attribute__((ext_vector_type(16))) __bf16        v16bf;
typedef __attribute__((ext_vector_type(8)))  float         v8f;
typedef __attribute__((ext_vector_type(4)))  unsigned int  v4u;
typedef __attribute__((ext_vector_type(8)))  int           v8i;
typedef __attribute__((ext_vector_type(4)))  int           v4i;

constexpr int B_ = 4096;
constexpr int O_ = 512;
constexpr int A_ = 64;
constexpr int H_ = 2048;
constexpr int Z_ = 256;
constexpr float CLAMP_ = 10.0f;

// ---------------------------------------------------------------------------
// TDM: 2D tile load global->LDS.  D# per CDNA5 ISA 8.3-8.6.
//   element = 2 bytes (bf16), tile = tile_w x tile_h, row stride in elements.
//   LDS pad: +16B after every 64B row chunk -> LDS row pitch 80B (LDA=40).
// ---------------------------------------------------------------------------
__device__ __forceinline__ void tdm_load_tile_2d(
    const __bf16* gtile, unsigned lds_off, unsigned tensor_w_elems,
    unsigned tensor_h_rows, unsigned row_stride_elems,
    unsigned tile_w, unsigned tile_h) {
  const unsigned long long ga = (unsigned long long)(uintptr_t)gtile;

  v4u g0;
  g0[0] = 1u;                                          // count=1 (valid user D#)
  g0[1] = lds_off;                                     // lds_addr (bytes)
  g0[2] = (unsigned)(ga & 0xffffffffu);                // global_addr[31:0]
  g0[3] = (unsigned)((ga >> 32) & 0x01ffffffu) |       // global_addr[56:32]
          (2u << 30);                                  // type = 2 ("image")

  v8i g1;
  g1[0] = (int)((1u << 16)        // data_size = 1 -> 2 bytes
              | (1u << 20)        // pad_enable
              | (3u << 22)        // pad_interval code 3 -> every 16 DWORDs (64B)
              | (3u << 25));      // pad_amount  code 3 -> 4 DWORDs (16B)
  g1[1] = (int)((tensor_w_elems & 0xffffu) << 16);               // dim0[15:0]
  g1[2] = (int)(((tensor_w_elems >> 16) & 0xffffu) |
                ((tensor_h_rows & 0xffffu) << 16));              // dim0[31:16] | dim1[15:0]
  g1[3] = (int)(((tensor_h_rows >> 16) & 0xffffu) |
                ((tile_w & 0xffffu) << 16));                     // dim1[31:16] | tile_dim0
  g1[4] = (int)(tile_h & 0xffffu);                               // tile_dim1 (tile_dim2=0)
  g1[5] = (int)row_stride_elems;                                 // dim0_stride[31:0]
  g1[6] = 0;                                                     // dim0_stride[47:32]
  g1[7] = 0;

  v4i g2 = {0, 0, 0, 0};
  v4i g3 = {0, 0, 0, 0};
#if defined(__clang_major__) && (__clang_major__ >= 23)
  v8i g4 = {0, 0, 0, 0, 0, 0, 0, 0};
  __builtin_amdgcn_tensor_load_to_lds(g0, g1, g2, g3, g4, 0);
#else
  __builtin_amdgcn_tensor_load_to_lds(g0, g1, g2, g3, 0);
#endif
}

// ---------------------------------------------------------------------------
// GEMM:  C[M,N] = act( A[M,K] @ W[N,K]^T + bias[N] ),  A/W bf16, acc fp32.
// A may be split into up to 3 row-major bf16 segments along K (concats).
// Block tile 128x128, 8 waves as 4(M) x 2(N); each wave 32x64 = 8 WMMA frags.
// Tiles staged by TDM, double buffered.
// ---------------------------------------------------------------------------
template <int ACT, int OUTB>  // ACT: relu?  OUTB: 1 = bf16 output, 0 = fp32
__global__ __launch_bounds__(256) void gemm_bf16_kernel(
    const __bf16* __restrict__ A0, int K0,
    const __bf16* __restrict__ A1, int K1,
    const __bf16* __restrict__ A2, int K2,
    const __bf16* __restrict__ W, const float* __restrict__ bias,
    void* __restrict__ Cout, int N, int Ktot) {
  constexpr int LDA = 40;  // bf16 elems per padded LDS row (64B data + 16B pad)
  __shared__ __bf16 lsA[2][128 * LDA];
  __shared__ __bf16 lsW[2][128 * LDA];

  const int tid  = threadIdx.x;
  const int lane = tid & 31;
  const int wave = tid >> 5;
  const int waveM = wave >> 1;  // 0..3 -> 32-row band
  const int waveN = wave & 1;   // 0..1 -> 64-col band
  const int l16  = lane & 15;
  const int half = lane >> 4;

  const int blockM = blockIdx.y * 128;
  const int blockN = blockIdx.x * 128;

  const unsigned ldsA[2] = {(unsigned)(uintptr_t)&lsA[0][0],
                            (unsigned)(uintptr_t)&lsA[1][0]};
  const unsigned ldsW[2] = {(unsigned)(uintptr_t)&lsW[0][0],
                            (unsigned)(uintptr_t)&lsW[1][0]};

  v8f zero = {};
  v8f acc[2][4];
#pragma unroll
  for (int ms = 0; ms < 2; ++ms)
#pragma unroll
    for (int ns = 0; ns < 4; ++ns) acc[ms][ns] = zero;

  const int T = Ktot / 32;

  // wave 0 issues the A tile, wave 1 the W tile (TDM ignores EXEC; one lane's
  // worth of scalar work per wave).
  auto issue = [&](int it, int buf) {
    const int k0 = it * 32;
    if (wave == 0) {
      const __bf16* Aseg; int segK, kk;
      if (k0 < K0)           { Aseg = A0; segK = K0; kk = k0; }
      else if (k0 < K0 + K1) { Aseg = A1; segK = K1; kk = k0 - K0; }
      else                   { Aseg = A2; segK = K2; kk = k0 - K0 - K1; }
      tdm_load_tile_2d(Aseg + (size_t)blockM * segK + kk, ldsA[buf],
                       (unsigned)segK, (unsigned)B_, (unsigned)segK, 32u, 128u);
    } else if (wave == 1) {
      tdm_load_tile_2d(W + (size_t)blockN * Ktot + k0, ldsW[buf],
                       (unsigned)Ktot, (unsigned)N, (unsigned)Ktot, 32u, 128u);
    }
  };

  issue(0, 0);  // prologue

  for (int it = 0; it < T; ++it) {
    __builtin_amdgcn_s_wait_tensorcnt(0);  // issuing waves drain their TDM op
    __syncthreads();                       // publish LDS tile to all waves
    if (it + 1 < T) issue(it + 1, (it + 1) & 1);  // prefetch next tile

    const int cb = it & 1;
    const __bf16* aB = &lsA[cb][0];
    const __bf16* wB = &lsW[cb][0];

    // Fragments, this code's k = 16*half + s for A and B (A: M=lane%16, K 16-per-halfwave;
    //                           B: N=lane%16, K 16-per-halfwave).
    v16bf afrag[2], bfrag[4];
#pragma unroll
    for (int ms = 0; ms < 2; ++ms) {
      const __bf16* src = aB + (waveM * 32 + ms * 16 + l16) * LDA;
#pragma unroll
      for (int i = 0; i < 8; ++i) {
        const int kb = 16 * half + 2 * i;
        afrag[ms][2 * i]     = src[kb];
        afrag[ms][2 * i + 1] = src[kb + 1];
      }
    }
#pragma unroll
    for (int ns = 0; ns < 4; ++ns) {
      const __bf16* src = wB + (waveN * 64 + ns * 16 + l16) * LDA;
#pragma unroll
      for (int i = 0; i < 8; ++i) {
        const int kb = 16 * half + 2 * i;
        bfrag[ns][2 * i]     = src[kb];
        bfrag[ns][2 * i + 1] = src[kb + 1];
      }
    }

#pragma unroll
    for (int ms = 0; ms < 2; ++ms)
#pragma unroll
      for (int ns = 0; ns < 4; ++ns)
        acc[ms][ns] = __builtin_amdgcn_wmma_f32_16x16x32_bf16(
            false, afrag[ms], false, bfrag[ns], (short)0, acc[ms][ns], false, false);

    __syncthreads();  // all waves done with buffer cb before TDM reuses it
  }

  // Epilogue (C/D layout: col = lane%16, row = v + 8*(lane/16))
#pragma unroll
  for (int ms = 0; ms < 2; ++ms) {
#pragma unroll
    for (int ns = 0; ns < 4; ++ns) {
      const int col = blockN + waveN * 64 + ns * 16 + l16;
      const float bv = bias[col];
#pragma unroll
      for (int v = 0; v < 8; ++v) {
        const int row = blockM + waveM * 32 + ms * 16 + half * 8 + v;
        float x = acc[ms][ns][v] + bv;
        if (ACT) x = fmaxf(x, 0.0f);
        if (OUTB) ((__bf16*)Cout)[(size_t)row * N + col] = (__bf16)x;
        else      ((float*)Cout)[(size_t)row * N + col] = x;
      }
    }
  }
}

// ---------------------------------------------------------------------------
// fp32 -> bf16 pack (4 elems / thread, b64 stores).  n must be mult of 4.
// ---------------------------------------------------------------------------
__global__ __launch_bounds__(256) void pack_bf16_kernel(
    const float* __restrict__ src, __bf16* __restrict__ dst, int n4) {
  const int i = blockIdx.x * 256 + threadIdx.x;
  if (i >= n4) return;
  const float4 v = ((const float4*)src)[i];
  union { __bf16 h[4]; unsigned long long u; } pk;
  pk.h[0] = (__bf16)v.x; pk.h[1] = (__bf16)v.y;
  pk.h[2] = (__bf16)v.z; pk.h[3] = (__bf16)v.w;
  ((unsigned long long*)dst)[i] = pk.u;
}

// ---------------------------------------------------------------------------
// Reparameterization: mu_q / clamped logvar_q / z_t (fp32 out + bf16 ws).
// ---------------------------------------------------------------------------
__global__ __launch_bounds__(256) void reparam_kernel(
    const float* __restrict__ enc_stats, const float* __restrict__ eps,
    float* __restrict__ mu_q, float* __restrict__ logvar_q,
    float* __restrict__ z_out, __bf16* __restrict__ z_bf) {
  const int idx = blockIdx.x * 256 + threadIdx.x;  // B*Z
  const int b = idx >> 8;   // Z_ == 256
  const int z = idx & 255;
  const float mu = enc_stats[(size_t)b * (2 * Z_) + z];
  float lv = enc_stats[(size_t)b * (2 * Z_) + Z_ + z];
  lv = fminf(fmaxf(lv, -CLAMP_), CLAMP_);
  const float zt = mu + eps[idx] * expf(0.5f * lv);
  mu_q[idx] = mu;
  logvar_q[idx] = lv;
  z_out[idx] = zt;
  z_bf[idx] = (__bf16)zt;
}

// ---------------------------------------------------------------------------
// Per-row NLL (over O) and KL (over Z).  One 256-thread block per batch row.
// ---------------------------------------------------------------------------
__global__ __launch_bounds__(256) void loss_kernel(
    const float* __restrict__ x, const float* __restrict__ dec_out,
    const float* __restrict__ prior_stats, const float* __restrict__ mu_q,
    const float* __restrict__ logvar_q, float* __restrict__ nll,
    float* __restrict__ kl) {
  __shared__ float red[256];
  const int b = blockIdx.x;
  const int t = threadIdx.x;

  float s = 0.0f;
#pragma unroll
  for (int o = t; o < O_; o += 256) {
    const float mu_x = dec_out[(size_t)b * (2 * O_) + o];
    float lvx = dec_out[(size_t)b * (2 * O_) + O_ + o];
    lvx = fminf(fmaxf(lvx, -CLAMP_), CLAMP_);
    const float d = x[(size_t)b * O_ + o] - mu_x;
    s += d * d * expf(-lvx) + lvx;
  }
  red[t] = s;
  __syncthreads();
  for (int st = 128; st > 0; st >>= 1) {
    if (t < st) red[t] += red[t + st];
    __syncthreads();
  }
  if (t == 0) nll[b] = 0.5f * red[0];
  __syncthreads();

  const float mu_p = prior_stats[(size_t)b * (2 * Z_) + t];
  const float lvp  = prior_stats[(size_t)b * (2 * Z_) + Z_ + t];
  const float mq   = mu_q[(size_t)b * Z_ + t];
  const float lq   = logvar_q[(size_t)b * Z_ + t];
  const float ivp  = expf(-lvp);
  const float dmu  = mq - mu_p;
  red[t] = dmu * dmu * ivp + expf(lq) * ivp + (lvp - lq) - 1.0f;
  __syncthreads();
  for (int st = 128; st > 0; st >>= 1) {
    if (t < st) red[t] += red[t + st];
    __syncthreads();
  }
  if (t == 0) kl[b] = 0.5f * red[0];
}

// ---------------------------------------------------------------------------
// GRU gate fusion (torch gate order r, z, n).
// ---------------------------------------------------------------------------
__global__ __launch_bounds__(256) void gru_kernel(
    const float* __restrict__ gi, const float* __restrict__ gh,
    const float* __restrict__ hprev, float* __restrict__ ht) {
  const size_t idx = (size_t)blockIdx.x * 256 + threadIdx.x;  // B*H
  const size_t b = idx / H_;
  const size_t h = idx % H_;
  const size_t base = b * (size_t)(3 * H_) + h;
  const float ir = gi[base], iz = gi[base + H_], in_ = gi[base + 2 * H_];
  const float hr = gh[base], hz = gh[base + H_], hn = gh[base + 2 * H_];
  const float r  = 1.0f / (1.0f + expf(-(ir + hr)));
  const float zg = 1.0f / (1.0f + expf(-(iz + hz)));
  const float n  = tanhf(in_ + r * hn);
  ht[idx] = (1.0f - zg) * n + zg * hprev[idx];
}

// ---------------------------------------------------------------------------
extern "C" void kernel_launch(void* const* d_in, const int* in_sizes, int n_in,
                              void* d_out, int out_size, void* d_ws, size_t ws_size,
                              hipStream_t stream) {
  const float* x_t      = (const float*)d_in[0];
  const float* a_prev   = (const float*)d_in[1];
  const float* h_prev   = (const float*)d_in[2];
  const float* eps      = (const float*)d_in[3];
  const float* prior_w1 = (const float*)d_in[4];
  const float* prior_b1 = (const float*)d_in[5];
  const float* prior_w2 = (const float*)d_in[6];
  const float* prior_b2 = (const float*)d_in[7];
  const float* enc_w1   = (const float*)d_in[8];
  const float* enc_b1   = (const float*)d_in[9];
  const float* enc_w2   = (const float*)d_in[10];
  const float* enc_b2   = (const float*)d_in[11];
  const float* dec_w1   = (const float*)d_in[12];
  const float* dec_b1   = (const float*)d_in[13];
  const float* dec_w2   = (const float*)d_in[14];
  const float* dec_b2   = (const float*)d_in[15];
  const float* w_ih     = (const float*)d_in[16];
  const float* w_hh     = (const float*)d_in[17];
  const float* b_ih     = (const float*)d_in[18];
  const float* b_hh     = (const float*)d_in[19];

  // ---- workspace layout ----
  char* w = (char*)d_ws;
  float* prior_stats = (float*)w;  w += (size_t)B_ * 2 * Z_ * 4;
  float* enc_stats   = (float*)w;  w += (size_t)B_ * 2 * Z_ * 4;
  float* dec_out     = (float*)w;  w += (size_t)B_ * 2 * O_ * 4;
  float* gi          = (float*)w;  w += (size_t)B_ * 3 * H_ * 4;
  float* gh          = (float*)w;  w += (size_t)B_ * 3 * H_ * 4;
  __bf16* h1b  = (__bf16*)w;  w += (size_t)B_ * H_ * 2;
  __bf16* xb   = (__bf16*)w;  w += (size_t)B_ * O_ * 2;
  __bf16* ab   = (__bf16*)w;  w += (size_t)B_ * A_ * 2;
  __bf16* hb   = (__bf16*)w;  w += (size_t)B_ * H_ * 2;
  __bf16* zb   = (__bf16*)w;  w += (size_t)B_ * Z_ * 2;
  __bf16* pw1b = (__bf16*)w;  w += (size_t)H_ * H_ * 2;
  __bf16* pw2b = (__bf16*)w;  w += (size_t)2 * Z_ * H_ * 2;
  __bf16* ew1b = (__bf16*)w;  w += (size_t)H_ * (O_ + H_) * 2;
  __bf16* ew2b = (__bf16*)w;  w += (size_t)2 * Z_ * H_ * 2;
  __bf16* dw1b = (__bf16*)w;  w += (size_t)H_ * (Z_ + H_) * 2;
  __bf16* dw2b = (__bf16*)w;  w += (size_t)2 * O_ * H_ * 2;
  __bf16* wihb = (__bf16*)w;  w += (size_t)3 * H_ * (O_ + Z_ + A_) * 2;
  __bf16* whhb = (__bf16*)w;  w += (size_t)3 * H_ * H_ * 2;

  // ---- output layout: (h_t, nll, kl, z_t, mu_q, logvar_q) ----
  float* out      = (float*)d_out;
  float* h_t      = out;
  float* nll      = out + (size_t)B_ * H_;
  float* kl       = nll + B_;
  float* z_t      = kl + B_;
  float* mu_q     = z_t + (size_t)B_ * Z_;
  float* logvar_q = mu_q + (size_t)B_ * Z_;

  const dim3 blk(256);
  const int gy = B_ / 128;  // 32

  auto pack = [&](const float* s, __bf16* d, size_t n) {
    const int n4 = (int)(n / 4);
    pack_bf16_kernel<<<dim3((n4 + 255) / 256), blk, 0, stream>>>(s, d, n4);
  };

  // 0) pre-pack everything the matrix cores touch to bf16
  pack(x_t, xb, (size_t)B_ * O_);
  pack(a_prev, ab, (size_t)B_ * A_);
  pack(h_prev, hb, (size_t)B_ * H_);
  pack(prior_w1, pw1b, (size_t)H_ * H_);
  pack(prior_w2, pw2b, (size_t)2 * Z_ * H_);
  pack(enc_w1, ew1b, (size_t)H_ * (O_ + H_));
  pack(enc_w2, ew2b, (size_t)2 * Z_ * H_);
  pack(dec_w1, dw1b, (size_t)H_ * (Z_ + H_));
  pack(dec_w2, dw2b, (size_t)2 * O_ * H_);
  pack(w_ih, wihb, (size_t)3 * H_ * (O_ + Z_ + A_));
  pack(w_hh, whhb, (size_t)3 * H_ * H_);

  // 1) prior hidden: relu(h @ prior_w1^T + b1) -> bf16       [B,H]
  gemm_bf16_kernel<1, 1><<<dim3(H_ / 128, gy), blk, 0, stream>>>(
      hb, H_, hb, 0, hb, 0, pw1b, prior_b1, h1b, H_, H_);
  // 2) prior stats -> fp32                                   [B,2Z]
  gemm_bf16_kernel<0, 0><<<dim3((2 * Z_) / 128, gy), blk, 0, stream>>>(
      h1b, H_, h1b, 0, h1b, 0, pw2b, prior_b2, prior_stats, 2 * Z_, H_);
  // 3) enc hidden: relu([x,h] @ enc_w1^T + b1) -> bf16       [B,H]
  gemm_bf16_kernel<1, 1><<<dim3(H_ / 128, gy), blk, 0, stream>>>(
      xb, O_, hb, H_, hb, 0, ew1b, enc_b1, h1b, H_, O_ + H_);
  // 4) enc stats -> fp32                                     [B,2Z]
  gemm_bf16_kernel<0, 0><<<dim3((2 * Z_) / 128, gy), blk, 0, stream>>>(
      h1b, H_, h1b, 0, h1b, 0, ew2b, enc_b2, enc_stats, 2 * Z_, H_);
  // 5) reparameterize -> mu_q, logvar_q, z_t (+ bf16 z)
  reparam_kernel<<<dim3((B_ * Z_) / 256), blk, 0, stream>>>(
      enc_stats, eps, mu_q, logvar_q, z_t, zb);
  // 6) dec hidden: relu([z,h] @ dec_w1^T + b1) -> bf16       [B,H]
  gemm_bf16_kernel<1, 1><<<dim3(H_ / 128, gy), blk, 0, stream>>>(
      zb, Z_, hb, H_, hb, 0, dw1b, dec_b1, h1b, H_, Z_ + H_);
  // 7) dec stats -> fp32                                     [B,2O]
  gemm_bf16_kernel<0, 0><<<dim3((2 * O_) / 128, gy), blk, 0, stream>>>(
      h1b, H_, h1b, 0, h1b, 0, dw2b, dec_b2, dec_out, 2 * O_, H_);
  // 8) GRU input gates: [x,z,a] @ w_ih^T + b_ih -> fp32      [B,3H]
  gemm_bf16_kernel<0, 0><<<dim3((3 * H_) / 128, gy), blk, 0, stream>>>(
      xb, O_, zb, Z_, ab, A_, wihb, b_ih, gi, 3 * H_, O_ + Z_ + A_);
  // 9) GRU hidden gates: h @ w_hh^T + b_hh -> fp32           [B,3H]
  gemm_bf16_kernel<0, 0><<<dim3((3 * H_) / 128, gy), blk, 0, stream>>>(
      hb, H_, hb, 0, hb, 0, whhb, b_hh, gh, 3 * H_, H_);
  // 10) NLL + KL reductions
  loss_kernel<<<dim3(B_), blk, 0, stream>>>(
      x_t, dec_out, prior_stats, mu_q, logvar_q, nll, kl);
  // 11) GRU gate fusion -> h_t
  gru_kernel<<<dim3((B_ * H_) / 256), blk, 0, stream>>>(gi, gh, h_prev, h_t);
}
